// ASAP_5111011083137
// MI455X (gfx1250) — matrix operand order repaired and checked by reading, not compiled, not run on hardware
//
#include <hip/hip_runtime.h>
#include <math.h>

// ---------------- problem constants (match reference) ----------------
#define C_HID   256
#define C_IN    128
#define C_OUT   8
#define C_NPER  512
#define C_NG    8
#define C_NN    4096          // N_NODES
#define C_KKEEP 256
#define C_NSEL  2048          // N_GRAPHS * K_KEEP
#define C_E     65536         // edges
#define C_E2    69632         // edges + self loops
#define C_EPS   1e-5f
#define C_SLOPE 0.2f

typedef __attribute__((ext_vector_type(16))) _Float16 v16h;
typedef __attribute__((ext_vector_type(8)))  float    v8f;

// ---------------- device helpers ----------------
__device__ __forceinline__ void edge2(int e, const int* __restrict__ src,
                                      const int* __restrict__ dst, int& s, int& d) {
  if (e < C_E) { s = src[e]; d = dst[e]; } else { s = e - C_E; d = s; }
}

__device__ __forceinline__ void atomic_max_f(float* addr, float val) {
  int* ai = (int*)addr;
  int cur = __float_as_int(*addr);
  while (__int_as_float(cur) < val) {
    int prev = atomicCAS(ai, cur, __float_as_int(val));
    if (prev == cur) break;
    cur = prev;
  }
}

// ---------------- generic WMMA f16->f32 GEMM ----------------
// C[M,N] = epilogue( (rowscale(A) * A)[M,K] @ B[K,N] )
// A element (m,k) at A[m*a_rs + k*a_cs]   (AC: a_cs==1, rows contiguous)
// B element (k,n) at B[k*b_ks + n*b_ns]   (BC: b_ks==1, k-runs contiguous; W[N,K] row-major)
// Out-of-range lanes clamp to a valid row/col (their outputs are never written),
// so all loads are unconditional -> vectorizable, no exec-mask churn.
template<bool AC, bool BC>
__global__ void k_wmma_gemm(const float* __restrict__ A, long a_rs, long a_cs,
                            const float* __restrict__ arowscale,
                            const float* __restrict__ B, long b_ks, long b_ns,
                            float* __restrict__ C, int M, int N, int K,
                            const float* __restrict__ bias,
                            const float* __restrict__ Cin,
                            const float* __restrict__ bg, const float* __restrict__ bb,
                            const float* __restrict__ brm, const float* __restrict__ brv,
                            int relu)
{
  int wave = blockIdx.x * (blockDim.x >> 5) + (threadIdx.x >> 5);
  int lane = threadIdx.x & 31;
  int tilesN = (N + 15) >> 4;
  int tilesM = (M + 15) >> 4;
  if (wave >= tilesM * tilesN) return;          // wave-uniform exit
  int tm = wave / tilesN;
  int tn = wave - tm * tilesN;
  int hs = lane >> 4;                           // lane half-select
  int r  = lane & 15;
  int m = tm * 16 + r;                          // A fragment row for this lane
  int n = tn * 16 + r;                          // B fragment col for this lane
  int mc = m < M ? m : M - 1;                   // clamped (safe: row not written)
  int nc = n < N ? n : N - 1;
  float asc = arowscale ? arowscale[mc] : 1.0f;
  const float* Ap = A + (long)mc * a_rs;
  const float* Bp = B + (long)nc * b_ns;

  v8f acc = {0.f,0.f,0.f,0.f,0.f,0.f,0.f,0.f};
  for (int kk = 0; kk < K; kk += 32) {
    int ka0 = kk + hs * 8;                      // A: K {0..7 | 8..15}
    int ka1 = ka0 + 16;                         // A: K {16..23 | 24..31}
    int kb  = kk + hs * 8;                      // B: K {0..7,16..23 | 8..15,24..31}
    float a0[8], a1[8], bt[16];
    if (AC) {
      float4 u0 = *(const float4*)(Ap + ka0);
      float4 u1 = *(const float4*)(Ap + ka0 + 4);
      float4 u2 = *(const float4*)(Ap + ka1);
      float4 u3 = *(const float4*)(Ap + ka1 + 4);
      a0[0]=u0.x; a0[1]=u0.y; a0[2]=u0.z; a0[3]=u0.w;
      a0[4]=u1.x; a0[5]=u1.y; a0[6]=u1.z; a0[7]=u1.w;
      a1[0]=u2.x; a1[1]=u2.y; a1[2]=u2.z; a1[3]=u2.w;
      a1[4]=u3.x; a1[5]=u3.y; a1[6]=u3.z; a1[7]=u3.w;
    } else {
#pragma unroll
      for (int i = 0; i < 8; ++i) {
        a0[i] = Ap[(long)(ka0 + i) * a_cs];
        a1[i] = Ap[(long)(ka1 + i) * a_cs];
      }
    }
    if (BC) {
      float4 w0 = *(const float4*)(Bp + kb);
      float4 w1 = *(const float4*)(Bp + kb + 4);
      float4 w2 = *(const float4*)(Bp + kb + 16);
      float4 w3 = *(const float4*)(Bp + kb + 20);
      bt[0]=w0.x;  bt[1]=w0.y;  bt[2]=w0.z;  bt[3]=w0.w;
      bt[4]=w1.x;  bt[5]=w1.y;  bt[6]=w1.z;  bt[7]=w1.w;
      bt[8]=w2.x;  bt[9]=w2.y;  bt[10]=w2.z; bt[11]=w2.w;
      bt[12]=w3.x; bt[13]=w3.y; bt[14]=w3.z; bt[15]=w3.w;
    } else {
#pragma unroll
      for (int i = 0; i < 16; ++i) bt[i] = Bp[(long)(kb + i + (i >= 8 ? 8 : 0)) * b_ks];
    }
    v16h af, bf;
#pragma unroll
    for (int i = 0; i < 8; ++i) {
      af[i]     = (_Float16)(a0[i] * asc);
      af[i + 8] = (_Float16)(a1[i] * asc);
    }
#pragma unroll
    for (int i = 0; i < 16; ++i) bf[i] = (_Float16)bt[i];

    acc = __builtin_amdgcn_wmma_f32_16x16x32_f16(false, af, false, bf,
                                                 (short)0, acc, false, false);
  }
  if (n < N) {
#pragma unroll
    for (int i = 0; i < 8; ++i) {
      int row = tm * 16 + hs * 8 + i;           // D layout: vgpr i -> M=i(+8)
      if (row < M) {
        float v = acc[i];
        if (bias) v += bias[n];
        if (Cin)  v += Cin[(long)row * N + n];
        if (bg)   v = (v - brm[n]) * bg[n] * rsqrtf(brv[n] + C_EPS) + bb[n];
        if (relu) v = v > 0.f ? v : 0.f;
        C[(long)row * N + n] = v;
      }
    }
  }
}

// ---------------- utility kernels ----------------
__global__ void k_zero(float* p, long n) {
  long i = (long)blockIdx.x * blockDim.x + threadIdx.x;
  if (i < n) p[i] = 0.f;
}
__global__ void k_fill(float* p, long n, float v) {
  long i = (long)blockIdx.x * blockDim.x + threadIdx.x;
  if (i < n) p[i] = v;
}
__global__ void k_ifill(int* p, long n, int v) {
  long i = (long)blockIdx.x * blockDim.x + threadIdx.x;
  if (i < n) p[i] = v;
}
__global__ void k_copy(float* d, const float* s, long n) {
  long i = (long)blockIdx.x * blockDim.x + threadIdx.x;
  if (i < n) d[i] = s[i];
}

// in-degree counts over real edges
__global__ void k_edge_count(const int* __restrict__ dst, float* cnt) {
  int e = blockIdx.x * blockDim.x + threadIdx.x;
  if (e < C_E) atomicAdd(&cnt[dst[e]], 1.0f);
}
__global__ void k_rowscale(const float* cnt, float* rs, int n) {
  int i = blockIdx.x * blockDim.x + threadIdx.x;
  if (i < n) rs[i] = 1.0f / fmaxf(cnt[i], 1.0f);
}

// agg[dst] += x[src]  (per edge, per feature)
__global__ void k_edge_scatter(const int* __restrict__ src, const int* __restrict__ dst,
                               const float* __restrict__ x, int H, float* __restrict__ agg) {
  long idx = (long)blockIdx.x * blockDim.x + threadIdx.x;
  if (idx >= (long)C_E * H) return;
  int e = (int)(idx / H), h = (int)(idx % H);
  atomicAdd(&agg[(long)dst[e] * H + h], x[(long)src[e] * H + h]);
}

// xq_raw[dst] = max(xq_raw[dst], x[src]) per feature
__global__ void k_edge_max(const int* __restrict__ src, const int* __restrict__ dst,
                           const float* __restrict__ x, int H, float* __restrict__ xq) {
  long idx = (long)blockIdx.x * blockDim.x + threadIdx.x;
  if (idx >= (long)C_E * H) return;
  int e = (int)(idx / H), h = (int)(idx % H);
  atomic_max_f(&xq[(long)dst[e] * H + h], x[(long)src[e] * H + h]);
}

// graph mean pool into h[g, coloff + c] (contiguous segments of `per` rows)
__global__ void k_gmp(const float* __restrict__ x, int per, int H,
                      float* __restrict__ hout, int ldh, int coloff) {
  int idx = blockIdx.x * blockDim.x + threadIdx.x;
  if (idx >= C_NG * H) return;
  int g = idx / H, c = idx % H;
  const float* base = x + (long)g * per * H + c;
  float s = 0.f;
  for (int i = 0; i < per; ++i) s += base[(long)i * H];
  hout[(long)g * ldh + coloff + c] = s / (float)per;
}

// out[i] = dot(x[i], w) (+ b[0])
__global__ void k_node_dot(const float* __restrict__ x, int H, const float* __restrict__ w,
                           const float* __restrict__ b, float* __restrict__ out, int n) {
  int i = blockIdx.x * blockDim.x + threadIdx.x;
  if (i >= n) return;
  const float* r = x + (long)i * H;
  float s = 0.f;
  for (int h = 0; h < H; ++h) s += r[h] * w[h];
  if (b) s += b[0];
  out[i] = s;
}

// attention logits with leaky relu + segment max
__global__ void k_att_sc(const float* __restrict__ qn, const float* __restrict__ jn,
                         const float* __restrict__ attb, const int* __restrict__ src,
                         const int* __restrict__ dst, float* __restrict__ sc,
                         float* __restrict__ mmax) {
  int e = blockIdx.x * blockDim.x + threadIdx.x;
  if (e >= C_E2) return;
  int s, d; edge2(e, src, dst, s, d);
  float v = qn[d] + jn[s] + attb[0];
  v = v >= 0.f ? v : C_SLOPE * v;
  sc[e] = v;
  atomic_max_f(&mmax[d], v);
}
__global__ void k_att_exp(float* __restrict__ sc, const float* __restrict__ mmax,
                          const int* __restrict__ src, const int* __restrict__ dst,
                          float* __restrict__ denom) {
  int e = blockIdx.x * blockDim.x + threadIdx.x;
  if (e >= C_E2) return;
  int s, d; edge2(e, src, dst, s, d);
  float v = expf(sc[e] - mmax[d]);
  sc[e] = v;
  atomicAdd(&denom[d], v);
}
__global__ void k_att_norm(const float* __restrict__ sc, const float* __restrict__ denom,
                           const int* __restrict__ src, const int* __restrict__ dst,
                           float* __restrict__ score) {
  int e = blockIdx.x * blockDim.x + threadIdx.x;
  if (e >= C_E2) return;
  int s, d; edge2(e, src, dst, s, d);
  score[e] = sc[e] / denom[d];
}

// xc[dst] += x[src] * score[e]
__global__ void k_edge_xc(const int* __restrict__ src, const int* __restrict__ dst,
                          const float* __restrict__ x, const float* __restrict__ score,
                          float* __restrict__ xc) {
  long idx = (long)blockIdx.x * blockDim.x + threadIdx.x;
  if (idx >= (long)C_E2 * C_HID) return;
  int e = (int)(idx / C_HID), h = (int)(idx % C_HID);
  int s, d; edge2(e, src, dst, s, d);
  atomicAdd(&xc[(long)d * C_HID + h], x[(long)s * C_HID + h] * score[e]);
}

// fitness aggregation: aggn[dst] += a[dst] - b[src]
__global__ void k_fitagg(const float* __restrict__ an, const float* __restrict__ bns,
                         const int* __restrict__ src, const int* __restrict__ dst,
                         float* __restrict__ aggn) {
  int e = blockIdx.x * blockDim.x + threadIdx.x;
  if (e >= C_E2) return;
  int s, d; edge2(e, src, dst, s, d);
  atomicAdd(&aggn[d], an[d] - bns[s]);
}
__global__ void k_fit(const float* __restrict__ aggn, const float* __restrict__ xc,
                      const float* __restrict__ w, const float* __restrict__ b,
                      float* __restrict__ fit) {
  int i = blockIdx.x * blockDim.x + threadIdx.x;
  if (i >= C_NN) return;
  float s = aggn[i];
  const float* r = xc + (long)i * C_HID;
  for (int h = 0; h < C_HID; ++h) s += r[h] * w[h];
  s += b[0];
  fit[i] = 1.f / (1.f + expf(-s));
}

// exact top-k per graph (rank = jax top_k order: desc, ties by index)
__global__ void k_topk(const float* __restrict__ fit, int* __restrict__ perm,
                       int* __restrict__ colOf) {
  int g = blockIdx.x, j = threadIdx.x;     // 512 threads
  __shared__ float f[C_NPER];
  f[j] = fit[g * C_NPER + j];
  __syncthreads();
  float fj = f[j];
  int rank = 0;
  for (int i = 0; i < C_NPER; ++i) {
    float fi = f[i];
    rank += (fi > fj) || (fi == fj && i < j);
  }
  if (rank < C_KKEEP) {
    int node = g * C_NPER + j;
    perm[g * C_KKEEP + rank] = node;
    colOf[node] = g * C_KKEEP + rank;
  }
}

__global__ void k_xnew(const float* __restrict__ xc, const float* __restrict__ fit,
                       const int* __restrict__ perm, float* __restrict__ xnew) {
  long idx = (long)blockIdx.x * blockDim.x + threadIdx.x;
  if (idx >= (long)C_NSEL * C_HID) return;
  int c = (int)(idx / C_HID), h = (int)(idx % C_HID);
  int node = perm[c];
  xnew[idx] = xc[(long)node * C_HID + h] * fit[node];
}

// dense S_sel[src, colOf[dst]] += score
__global__ void k_ssel(const int* __restrict__ src, const int* __restrict__ dst,
                       const int* __restrict__ colOf, const float* __restrict__ score,
                       float* __restrict__ Ssel) {
  int e = blockIdx.x * blockDim.x + threadIdx.x;
  if (e >= C_E2) return;
  int s, d; edge2(e, src, dst, s, d);
  int c = colOf[d];
  if (c >= 0) atomicAdd(&Ssel[(long)s * C_NSEL + c], score[e]);
}
// T = A @ S_sel : T[src, :] += S_sel[dst, :]
__global__ void k_T(const int* __restrict__ src, const int* __restrict__ dst,
                    const float* __restrict__ Ssel, float* __restrict__ T) {
  long idx = (long)blockIdx.x * blockDim.x + threadIdx.x;
  if (idx >= (long)C_E2 * C_NSEL) return;
  int e = (int)(idx / C_NSEL), c = (int)(idx % C_NSEL);
  int s, d; edge2(e, src, dst, s, d);
  float v = Ssel[(long)d * C_NSEL + c];
  if (v != 0.f) atomicAdd(&T[(long)s * C_NSEL + c], v);
}
// A2[colOf[dst], :] += score * T[src, :]
__global__ void k_A2(const int* __restrict__ src, const int* __restrict__ dst,
                     const int* __restrict__ colOf, const float* __restrict__ score,
                     const float* __restrict__ T, float* __restrict__ A2) {
  long idx = (long)blockIdx.x * blockDim.x + threadIdx.x;
  if (idx >= (long)C_E2 * C_NSEL) return;
  int e = (int)(idx / C_NSEL), c = (int)(idx % C_NSEL);
  int s, d; edge2(e, src, dst, s, d);
  int cc = colOf[d];
  if (cc >= 0) {
    float v = score[e] * T[(long)s * C_NSEL + c];
    if (v != 0.f) atomicAdd(&A2[(long)cc * C_NSEL + c], v);
  }
}
__global__ void k_diag0(float* A2) {
  int i = blockIdx.x * blockDim.x + threadIdx.x;
  if (i < C_NSEL) A2[(long)i * C_NSEL + i] = 0.f;
}
// column non-zero counts of A2 -> reciprocal row scale for agg
__global__ void k_cnt2(const float* __restrict__ A2, float* __restrict__ rs2) {
  int j = blockIdx.x * blockDim.x + threadIdx.x;
  if (j >= C_NSEL) return;
  int c = 0;
  for (int i = 0; i < C_NSEL; ++i) c += (A2[(long)i * C_NSEL + j] != 0.f);
  rs2[j] = 1.f / fmaxf((float)c, 1.f);
}

// ---------------- host orchestration ----------------
extern "C" void kernel_launch(void* const* d_in, const int* in_sizes, int n_in,
                              void* d_out, int out_size, void* d_ws, size_t ws_size,
                              hipStream_t stream)
{
  (void)in_sizes; (void)n_in; (void)out_size; (void)ws_size;
  const float* x   = (const float*)d_in[0];
  const int*   ei  = (const int*)d_in[1];
  const int* src = ei;
  const int* dst = ei + C_E;
  // params in setup_inputs() dict order
  int p = 3;
  const float* c1Wrel = (const float*)d_in[p++]; const float* c1brel = (const float*)d_in[p++]; const float* c1Wroot = (const float*)d_in[p++];
  const float* c2Wrel = (const float*)d_in[p++]; const float* c2brel = (const float*)d_in[p++]; const float* c2Wroot = (const float*)d_in[p++];
  const float* c3Wrel = (const float*)d_in[p++]; const float* c3brel = (const float*)d_in[p++]; const float* c3Wroot = (const float*)d_in[p++];
  const float* c4Wrel = (const float*)d_in[p++]; const float* c4brel = (const float*)d_in[p++]; const float* c4Wroot = (const float*)d_in[p++];
  const float* bn0g = (const float*)d_in[p++]; const float* bn0b = (const float*)d_in[p++]; const float* bn0m = (const float*)d_in[p++]; const float* bn0v = (const float*)d_in[p++];
  const float* bn1g = (const float*)d_in[p++]; const float* bn1b = (const float*)d_in[p++]; const float* bn1m = (const float*)d_in[p++]; const float* bn1v = (const float*)d_in[p++];
  const float* bn2g = (const float*)d_in[p++]; const float* bn2b = (const float*)d_in[p++]; const float* bn2m = (const float*)d_in[p++]; const float* bn2v = (const float*)d_in[p++];
  const float* linW = (const float*)d_in[p++]; const float* linb = (const float*)d_in[p++];
  const float* w_aq = (const float*)d_in[p++]; const float* w_aj = (const float*)d_in[p++];
  const float* attb = (const float*)d_in[p++];
  const float* le1W = (const float*)d_in[p++]; const float* le1b = (const float*)d_in[p++];
  const float* le2W = (const float*)d_in[p++];
  const float* le3W = (const float*)d_in[p++]; const float* le3b = (const float*)d_in[p++];
  const float* lin1W = (const float*)d_in[p++]; const float* lin1b = (const float*)d_in[p++];
  const float* lin2W = (const float*)d_in[p++]; const float* lin2b = (const float*)d_in[p++];

  // workspace bump allocator (256B aligned)
  char* wsb = (char*)d_ws;
  size_t off = 0;
  auto allocf = [&](size_t n) { float* q = (float*)(wsb + off); off += ((n * 4 + 255) / 256) * 256; return q; };
  auto alloci = [&](size_t n) { int*   q = (int*)(wsb + off);   off += ((n * 4 + 255) / 256) * 256; return q; };

  float* x1   = allocf((size_t)C_NN * C_HID);
  float* aggb = allocf((size_t)C_NN * C_HID);
  float* tmpb = allocf((size_t)C_NN * C_HID);
  float* x2   = allocf((size_t)C_NN * C_HID);
  float* xqr  = allocf((size_t)C_NN * C_HID);
  float* xq   = allocf((size_t)C_NN * C_HID);
  float* xc   = allocf((size_t)C_NN * C_HID);
  float* cnt  = allocf(C_NN);
  float* rs   = allocf(C_NN);
  float* qn   = allocf(C_NN);
  float* jn   = allocf(C_NN);
  float* an   = allocf(C_NN);
  float* bns  = allocf(C_NN);
  float* aggn = allocf(C_NN);
  float* fit  = allocf(C_NN);
  float* mmax = allocf(C_NN);
  float* denom= allocf(C_NN);
  float* sc   = allocf(C_E2);
  float* score= allocf(C_E2);
  int*   perm = alloci(C_NSEL);
  int*   colOf= alloci(C_NN);
  float* xnew = allocf((size_t)C_NSEL * C_HID);
  float* aggP = allocf((size_t)C_NSEL * C_HID);
  float* x3   = allocf((size_t)C_NSEL * C_HID);
  float* rs2  = allocf(C_NSEL);
  float* hb   = allocf((size_t)C_NG * 4 * C_HID);
  float* zb   = allocf((size_t)C_NG * C_HID);
  float* Ssel = allocf((size_t)C_NN * C_NSEL);
  float* Tb   = allocf((size_t)C_NN * C_NSEL);
  float* A2   = allocf((size_t)C_NSEL * C_NSEL);

  auto nb = [](long n) { return (unsigned)((n + 255) / 256); };
  auto zero = [&](float* q, long n) { k_zero<<<nb(n), 256, 0, stream>>>(q, n); };
  auto gemm = [&](bool contig, const float* A, long ars, long acs, const float* arsR,
                  const float* B, long bks, long bns2, float* C,
                  int M, int N, int K, const float* bias, const float* Cin,
                  const float* g, const float* b, const float* rm, const float* rv, int relu) {
    int tiles = ((M + 15) / 16) * ((N + 15) / 16);
    int blocks = (tiles + 7) / 8;                      // 8 wave32 per block
    if (contig)
      k_wmma_gemm<true, true><<<blocks, 256, 0, stream>>>(A, ars, acs, arsR, B, bks, bns2, C,
                                                          M, N, K, bias, Cin, g, b, rm, rv, relu);
    else
      k_wmma_gemm<false, false><<<blocks, 256, 0, stream>>>(A, ars, acs, arsR, B, bks, bns2, C,
                                                            M, N, K, bias, Cin, g, b, rm, rv, relu);
  };

  // ---- in-degree counts / mean scale (shared by conv1 & conv2) ----
  zero(cnt, C_NN);
  k_edge_count<<<nb(C_E), 256, 0, stream>>>(dst, cnt);
  k_rowscale<<<nb(C_NN), 256, 0, stream>>>(cnt, rs, C_NN);

  // ---- conv1 : relu(mean-agg @ Wrel^T + brel + x @ Wroot^T) ----
  zero(aggb, (long)C_NN * C_IN);
  k_edge_scatter<<<nb((long)C_E * C_IN), 256, 0, stream>>>(src, dst, x, C_IN, aggb);
  gemm(true, x,    C_IN, 1, nullptr, c1Wroot, 1, C_IN, tmpb, C_NN, C_HID, C_IN, nullptr, nullptr, nullptr, nullptr, nullptr, nullptr, 0);
  gemm(true, aggb, C_IN, 1, rs,      c1Wrel,  1, C_IN, x1,   C_NN, C_HID, C_IN, c1brel, tmpb,   nullptr, nullptr, nullptr, nullptr, 1);
  k_gmp<<<nb(C_NG * C_HID), 256, 0, stream>>>(x1, C_NPER, C_HID, hb, 4 * C_HID, 0);

  // ---- conv2 + bn0 + relu ----
  zero(aggb, (long)C_NN * C_HID);
  k_edge_scatter<<<nb((long)C_E * C_HID), 256, 0, stream>>>(src, dst, x1, C_HID, aggb);
  gemm(true, x1,   C_HID, 1, nullptr, c2Wroot, 1, C_HID, tmpb, C_NN, C_HID, C_HID, nullptr, nullptr, nullptr, nullptr, nullptr, nullptr, 0);
  gemm(true, aggb, C_HID, 1, rs,      c2Wrel,  1, C_HID, x2,   C_NN, C_HID, C_HID, c2brel, tmpb,   bn0g, bn0b, bn0m, bn0v, 1);
  k_gmp<<<nb(C_NG * C_HID), 256, 0, stream>>>(x2, C_NPER, C_HID, hb, 4 * C_HID, C_HID);

  // ---- ASAP pool ----
  k_copy<<<nb((long)C_NN * C_HID), 256, 0, stream>>>(xqr, x2, (long)C_NN * C_HID); // self loops
  k_edge_max<<<nb((long)C_E * C_HID), 256, 0, stream>>>(src, dst, x2, C_HID, xqr);
  gemm(true, xqr, C_HID, 1, nullptr, linW, 1, C_HID, xq, C_NN, C_HID, C_HID, linb, nullptr, nullptr, nullptr, nullptr, nullptr, 0);
  k_node_dot<<<nb(C_NN), 256, 0, stream>>>(xq, C_HID, w_aq, nullptr, qn, C_NN);
  k_node_dot<<<nb(C_NN), 256, 0, stream>>>(x2, C_HID, w_aj, nullptr, jn, C_NN);
  k_fill<<<nb(C_NN), 256, 0, stream>>>(mmax, C_NN, -1e30f);
  k_att_sc<<<nb(C_E2), 256, 0, stream>>>(qn, jn, attb, src, dst, sc, mmax);
  zero(denom, C_NN);
  k_att_exp<<<nb(C_E2), 256, 0, stream>>>(sc, mmax, src, dst, denom);
  k_att_norm<<<nb(C_E2), 256, 0, stream>>>(sc, denom, src, dst, score);
  zero(xc, (long)C_NN * C_HID);
  k_edge_xc<<<nb((long)C_E2 * C_HID), 256, 0, stream>>>(src, dst, x2, score, xc);
  k_node_dot<<<nb(C_NN), 256, 0, stream>>>(xc, C_HID, le1W, le1b, an, C_NN);
  k_node_dot<<<nb(C_NN), 256, 0, stream>>>(xc, C_HID, le2W, nullptr, bns, C_NN);
  zero(aggn, C_NN);
  k_fitagg<<<nb(C_E2), 256, 0, stream>>>(an, bns, src, dst, aggn);
  k_fit<<<nb(C_NN), 256, 0, stream>>>(aggn, xc, le3W, le3b, fit);
  k_ifill<<<nb(C_NN), 256, 0, stream>>>(colOf, C_NN, -1);
  k_topk<<<C_NG, C_NPER, 0, stream>>>(fit, perm, colOf);
  k_xnew<<<nb((long)C_NSEL * C_HID), 256, 0, stream>>>(xc, fit, perm, xnew);

  // A2 = S_sel^T @ (A @ S_sel), zero diagonal
  zero(Ssel, (long)C_NN * C_NSEL);
  k_ssel<<<nb(C_E2), 256, 0, stream>>>(src, dst, colOf, score, Ssel);
  zero(Tb, (long)C_NN * C_NSEL);
  k_T<<<nb((long)C_E2 * C_NSEL), 256, 0, stream>>>(src, dst, Ssel, Tb);
  zero(A2, (long)C_NSEL * C_NSEL);
  k_A2<<<nb((long)C_E2 * C_NSEL), 256, 0, stream>>>(src, dst, colOf, score, Tb, A2);
  k_diag0<<<nb(C_NSEL), 256, 0, stream>>>(A2);
  k_cnt2<<<nb(C_NSEL), 256, 0, stream>>>(A2, rs2);

  // ---- conv3 (dense) + bn1 + relu ----  agg = (A2^T @ xnew) / cnt2
  gemm(false, A2,   1, C_NSEL, rs2,    xnew,    C_HID, 1, aggP, C_NSEL, C_HID, C_NSEL, nullptr, nullptr, nullptr, nullptr, nullptr, nullptr, 0);
  gemm(true,  xnew, C_HID, 1, nullptr, c3Wroot, 1, C_HID,  tmpb, C_NSEL, C_HID, C_HID,  nullptr, nullptr, nullptr, nullptr, nullptr, nullptr, 0);
  gemm(true,  aggP, C_HID, 1, nullptr, c3Wrel,  1, C_HID,  x3,   C_NSEL, C_HID, C_HID,  c3brel,  tmpb,   bn1g, bn1b, bn1m, bn1v, 1);
  k_gmp<<<nb(C_NG * C_HID), 256, 0, stream>>>(x3, C_KKEEP, C_HID, hb, 4 * C_HID, 2 * C_HID);

  // ---- conv4 (dense) + bn2 + relu ----
  gemm(false, A2,  1, C_NSEL, rs2,    x3,      C_HID, 1, aggP, C_NSEL, C_HID, C_NSEL, nullptr, nullptr, nullptr, nullptr, nullptr, nullptr, 0);
  gemm(true,  x3,  C_HID, 1, nullptr, c4Wroot, 1, C_HID,  tmpb, C_NSEL, C_HID, C_HID,  nullptr, nullptr, nullptr, nullptr, nullptr, nullptr, 0);
  gemm(true,  aggP,C_HID, 1, nullptr, c4Wrel,  1, C_HID,  xnew, C_NSEL, C_HID, C_HID,  c4brel,  tmpb,   bn2g, bn2b, bn2m, bn2v, 1);
  k_gmp<<<nb(C_NG * C_HID), 256, 0, stream>>>(xnew, C_KKEEP, C_HID, hb, 4 * C_HID, 3 * C_HID);

  // ---- MLP head ----
  gemm(true, hb, 4 * C_HID, 1, nullptr, lin1W, 1, 4 * C_HID, zb, C_NG, C_HID, 4 * C_HID, lin1b, nullptr, nullptr, nullptr, nullptr, nullptr, 1);
  gemm(true, zb, C_HID,     1, nullptr, lin2W, 1, C_HID, (float*)d_out, C_NG, C_OUT, C_HID, lin2b, nullptr, nullptr, nullptr, nullptr, nullptr, 0);
}
